// ENGNLayer_54485955117752
// MI455X (gfx1250) — matrix operand order repaired and checked by reading, not compiled, not run on hardware
//
#include <hip/hip_runtime.h>

// ---------------------------------------------------------------------------
// ENGN layer for MI455X (gfx1250): bf16 WMMA GEMMs + f32 everything else.
// ---------------------------------------------------------------------------

typedef __bf16 v16bf __attribute__((ext_vector_type(16)));
typedef float  v8f   __attribute__((ext_vector_type(8)));

union FragB16 {
    v16bf v;
    uint4 q[2];
};

__device__ __forceinline__ unsigned short f32_to_bf16(float f) {
    unsigned int u = __float_as_uint(f);
    u += 0x7FFFu + ((u >> 16) & 1u);   // round to nearest even
    return (unsigned short)(u >> 16);
}

__device__ __forceinline__ float silu_f32(float x) {
    return x / (1.0f + __expf(-x));
}

// ---------------------------------------------------------------------------
// 1) LayerNorm: one wave (32 lanes) per 128-elem row; emit bf16 h.
// ---------------------------------------------------------------------------
__global__ __launch_bounds__(256) void ln_to_bf16_kernel(
    const float* __restrict__ x, const float* __restrict__ gamma,
    const float* __restrict__ beta, unsigned short* __restrict__ hbf, int N)
{
    int wave = threadIdx.x >> 5;
    int lane = threadIdx.x & 31;
    int row  = blockIdx.x * 8 + wave;
    if (row >= N) return;

    const float4 v = *(const float4*)(x + (size_t)row * 128 + lane * 4);
    float s  = v.x + v.y + v.z + v.w;
    float sq = v.x * v.x + v.y * v.y + v.z * v.z + v.w * v.w;
    #pragma unroll
    for (int o = 16; o > 0; o >>= 1) {
        s  += __shfl_xor(s,  o, 32);
        sq += __shfl_xor(sq, o, 32);
    }
    float mean = s * (1.0f / 128.0f);
    float var  = sq * (1.0f / 128.0f) - mean * mean;
    float inv  = rsqrtf(var + 1e-5f);

    const float4 g = *(const float4*)(gamma + lane * 4);
    const float4 b = *(const float4*)(beta  + lane * 4);
    unsigned short r0 = f32_to_bf16((v.x - mean) * inv * g.x + b.x);
    unsigned short r1 = f32_to_bf16((v.y - mean) * inv * g.y + b.y);
    unsigned short r2 = f32_to_bf16((v.z - mean) * inv * g.z + b.z);
    unsigned short r3 = f32_to_bf16((v.w - mean) * inv * g.w + b.w);
    uint2 p;
    p.x = (unsigned)r0 | ((unsigned)r1 << 16);
    p.y = (unsigned)r2 | ((unsigned)r3 << 16);
    *(uint2*)(hbf + (size_t)row * 128 + lane * 4) = p;
}

// ---------------------------------------------------------------------------
// 2) Swizzle f32 weights [K x 128] into per-lane WMMA-B bf16 layout:
//    out[(((kt*8 + nt)*32 + lane)*16 + j)] = W[kt*32 + 16*(j>>3) + 8*(lane>>4) + (j&7)][nt*16 + (lane&15)]
// ---------------------------------------------------------------------------
__global__ __launch_bounds__(256) void swizzle_w_kernel(
    const float* __restrict__ W, unsigned short* __restrict__ out, int Ktiles)
{
    int idx   = blockIdx.x * blockDim.x + threadIdx.x;
    int total = Ktiles * 8 * 32 * 16;
    if (idx >= total) return;
    int j    =  idx        & 15;
    int lane = (idx >> 4)  & 31;
    int nt   = (idx >> 9)  & 7;
    int kt   =  idx >> 12;
    int k = kt * 32 + (j >> 3) * 16 + (lane >> 4) * 8 + (j & 7);
    int n = nt * 16 + (lane & 15);
    out[idx] = f32_to_bf16(W[(size_t)k * 128 + n]);
}

// ---------------------------------------------------------------------------
// 3) Edge in-degree counts (by src)
// ---------------------------------------------------------------------------
__global__ __launch_bounds__(256) void edge_count_kernel(
    const int* __restrict__ src, float* __restrict__ cnt, int E)
{
    int i = blockIdx.x * blockDim.x + threadIdx.x;
    if (i < E) atomicAdd(&cnt[src[i]], 1.0f);
}

// ---------------------------------------------------------------------------
// 4) Edge message MLP: 16 edges per block, 8 waves = 8 N-tiles of 16 cols.
//    layer1: [16x384]@[384x128]  (12 WMMA/wave), SiLU
//    layer2: [16x128]@[128x128]  (4 WMMA/wave),  SiLU, f32 atomic scatter-add.
// ---------------------------------------------------------------------------
__global__ __launch_bounds__(256) void edge_msg_kernel(
    const unsigned short* __restrict__ hbf, const float* __restrict__ ef,
    const int* __restrict__ srcIdx, const int* __restrict__ dstIdx,
    const unsigned short* __restrict__ w1sw, const float* __restrict__ b1,
    const unsigned short* __restrict__ w2sw, const float* __restrict__ b2,
    float* __restrict__ msum, int E)
{
    __shared__ unsigned short smA[16][392];   // 384 + 8 pad: stride 784B = 196 dw ≡ 4 mod 64 banks
    __shared__ unsigned short smM[16][136];   // 128 + 8 pad: stride 272B = 68 dw  ≡ 4 mod 64 banks
    __shared__ int sSrc[16];

    const int tid = threadIdx.x;
    const int e0  = blockIdx.x * 16;

    if (tid < 16) {
        int e = e0 + tid;
        sSrc[tid] = (e < E) ? srcIdx[e] : -1;
    }

    // ---- stage A = [hi | hj | ef] as bf16, 16-byte chunks ----
    for (int chunk = tid; chunk < 16 * 48; chunk += 256) {
        int eL = chunk / 48;
        int cc = chunk % 48;
        int e  = e0 + eL;
        uint4 q = make_uint4(0u, 0u, 0u, 0u);
        if (e < E) {
            if (cc < 16) {
                int s = srcIdx[e];
                q = *(const uint4*)(hbf + (size_t)s * 128 + cc * 8);
            } else if (cc < 32) {
                int d = dstIdx[e];
                q = *(const uint4*)(hbf + (size_t)d * 128 + (cc - 16) * 8);
            } else {
                const float* fp = ef + (size_t)e * 128 + (cc - 32) * 8;
                float4 f0 = *(const float4*)(fp);
                float4 f1 = *(const float4*)(fp + 4);
                q.x = (unsigned)f32_to_bf16(f0.x) | ((unsigned)f32_to_bf16(f0.y) << 16);
                q.y = (unsigned)f32_to_bf16(f0.z) | ((unsigned)f32_to_bf16(f0.w) << 16);
                q.z = (unsigned)f32_to_bf16(f1.x) | ((unsigned)f32_to_bf16(f1.y) << 16);
                q.w = (unsigned)f32_to_bf16(f1.z) | ((unsigned)f32_to_bf16(f1.w) << 16);
            }
        }
        *(uint4*)(&smA[eL][cc * 8]) = q;
    }
    __syncthreads();

    const int w   = tid >> 5;       // N-tile (0..7)
    const int l   = tid & 31;       // lane
    const int rn  = l & 15;         // row for A, column-in-tile for C/D
    const int hi  = l >> 4;         // K-half selector
    const int col = w * 16 + rn;    // global output column

    // ---- layer 1: K = 384 ----
    v8f acc = {};
    #pragma unroll
    for (int kt = 0; kt < 12; ++kt) {
        FragB16 a, b;
        const unsigned short* ap = &smA[rn][kt * 32 + hi * 8];
        a.q[0] = *(const uint4*)(ap);
        a.q[1] = *(const uint4*)(ap + 16);
        const unsigned short* bp = w1sw + ((size_t)((kt * 8 + w) * 32 + l) << 4);
        b.q[0] = *(const uint4*)(bp);
        b.q[1] = *(const uint4*)(bp + 8);
        acc = __builtin_amdgcn_wmma_f32_16x16x32_bf16(
            false, a.v, false, b.v, (short)0, acc, false, false);
    }
    float bias1 = b1[col];
    #pragma unroll
    for (int r = 0; r < 8; ++r) {
        float y = silu_f32(acc[r] + bias1);
        smM[r + hi * 8][col] = f32_to_bf16(y);
    }
    __syncthreads();

    // ---- layer 2: K = 128 ----
    v8f acc2 = {};
    #pragma unroll
    for (int kt = 0; kt < 4; ++kt) {
        FragB16 a, b;
        const unsigned short* ap = &smM[rn][kt * 32 + hi * 8];
        a.q[0] = *(const uint4*)(ap);
        a.q[1] = *(const uint4*)(ap + 16);
        const unsigned short* bp = w2sw + ((size_t)((kt * 8 + w) * 32 + l) << 4);
        b.q[0] = *(const uint4*)(bp);
        b.q[1] = *(const uint4*)(bp + 8);
        acc2 = __builtin_amdgcn_wmma_f32_16x16x32_bf16(
            false, a.v, false, b.v, (short)0, acc2, false, false);
    }
    float bias2 = b2[col];
    #pragma unroll
    for (int r = 0; r < 8; ++r) {
        int eL = r + hi * 8;
        if (e0 + eL < E) {
            float m = silu_f32(acc2[r] + bias2);
            atomicAdd(&msum[(size_t)sSrc[eL] * 128 + col], m);
        }
    }
}

// ---------------------------------------------------------------------------
// 5) Node aggregation MLP + residual: 16 nodes per block, same wave layout.
//    layer1: [16x256]@[256x128] (8 WMMA/wave), SiLU
//    layer2: [16x128]@[128x128] (4 WMMA/wave), SiLU, +residual, f32 out.
// ---------------------------------------------------------------------------
__global__ __launch_bounds__(256) void node_agg_kernel(
    const float* __restrict__ x0, const unsigned short* __restrict__ hbf,
    const float* __restrict__ msum, const float* __restrict__ cnt,
    const unsigned short* __restrict__ w1sw, const float* __restrict__ b1,
    const unsigned short* __restrict__ w2sw, const float* __restrict__ b2,
    float* __restrict__ out, int N)
{
    __shared__ unsigned short smA[16][264];   // 256 + 8 pad (528B = 132 dw ≡ 4 mod 64)
    __shared__ unsigned short smM[16][136];
    __shared__ float sInv[16];

    const int tid = threadIdx.x;
    const int n0  = blockIdx.x * 16;

    if (tid < 16) {
        int n = n0 + tid;
        float c = (n < N) ? cnt[n] : 1.0f;
        sInv[tid] = 1.0f / fmaxf(c, 1.0f);
    }
    __syncthreads();

    // ---- stage A = [h | mean] as bf16 ----
    for (int chunk = tid; chunk < 16 * 32; chunk += 256) {
        int nL = chunk >> 5;
        int cc = chunk & 31;
        int n  = n0 + nL;
        uint4 q = make_uint4(0u, 0u, 0u, 0u);
        if (n < N) {
            if (cc < 16) {
                q = *(const uint4*)(hbf + (size_t)n * 128 + cc * 8);
            } else {
                const float* mp = msum + (size_t)n * 128 + (cc - 16) * 8;
                float inv = sInv[nL];
                float4 f0 = *(const float4*)(mp);
                float4 f1 = *(const float4*)(mp + 4);
                q.x = (unsigned)f32_to_bf16(f0.x * inv) | ((unsigned)f32_to_bf16(f0.y * inv) << 16);
                q.y = (unsigned)f32_to_bf16(f0.z * inv) | ((unsigned)f32_to_bf16(f0.w * inv) << 16);
                q.z = (unsigned)f32_to_bf16(f1.x * inv) | ((unsigned)f32_to_bf16(f1.y * inv) << 16);
                q.w = (unsigned)f32_to_bf16(f1.z * inv) | ((unsigned)f32_to_bf16(f1.w * inv) << 16);
            }
        }
        *(uint4*)(&smA[nL][cc * 8]) = q;
    }
    __syncthreads();

    const int w   = tid >> 5;
    const int l   = tid & 31;
    const int rn  = l & 15;
    const int hi  = l >> 4;
    const int col = w * 16 + rn;

    // ---- layer 1: K = 256 ----
    v8f acc = {};
    #pragma unroll
    for (int kt = 0; kt < 8; ++kt) {
        FragB16 a, b;
        const unsigned short* ap = &smA[rn][kt * 32 + hi * 8];
        a.q[0] = *(const uint4*)(ap);
        a.q[1] = *(const uint4*)(ap + 16);
        const unsigned short* bp = w1sw + ((size_t)((kt * 8 + w) * 32 + l) << 4);
        b.q[0] = *(const uint4*)(bp);
        b.q[1] = *(const uint4*)(bp + 8);
        acc = __builtin_amdgcn_wmma_f32_16x16x32_bf16(
            false, a.v, false, b.v, (short)0, acc, false, false);
    }
    float bias1 = b1[col];
    #pragma unroll
    for (int r = 0; r < 8; ++r) {
        float y = silu_f32(acc[r] + bias1);
        smM[r + hi * 8][col] = f32_to_bf16(y);
    }
    __syncthreads();

    // ---- layer 2: K = 128 ----
    v8f acc2 = {};
    #pragma unroll
    for (int kt = 0; kt < 4; ++kt) {
        FragB16 a, b;
        const unsigned short* ap = &smM[rn][kt * 32 + hi * 8];
        a.q[0] = *(const uint4*)(ap);
        a.q[1] = *(const uint4*)(ap + 16);
        const unsigned short* bp = w2sw + ((size_t)((kt * 8 + w) * 32 + l) << 4);
        b.q[0] = *(const uint4*)(bp);
        b.q[1] = *(const uint4*)(bp + 8);
        acc2 = __builtin_amdgcn_wmma_f32_16x16x32_bf16(
            false, a.v, false, b.v, (short)0, acc2, false, false);
    }
    float bias2 = b2[col];
    #pragma unroll
    for (int r = 0; r < 8; ++r) {
        int nL = r + hi * 8;
        int n  = n0 + nL;
        if (n < N) {
            float y = silu_f32(acc2[r] + bias2);
            out[(size_t)n * 128 + col] = x0[(size_t)n * 128 + col] + y;
        }
    }
}

// ---------------------------------------------------------------------------
// Host launcher
// ---------------------------------------------------------------------------
extern "C" void kernel_launch(void* const* d_in, const int* in_sizes, int n_in,
                              void* d_out, int out_size, void* d_ws, size_t ws_size,
                              hipStream_t stream)
{
    const float* node  = (const float*)d_in[0];
    const float* ef    = (const float*)d_in[1];
    const float* gamma = (const float*)d_in[2];
    const float* beta  = (const float*)d_in[3];
    const float* w1    = (const float*)d_in[4];
    const float* b1    = (const float*)d_in[5];
    const float* w2    = (const float*)d_in[6];
    const float* b2    = (const float*)d_in[7];
    const float* aw1   = (const float*)d_in[8];
    const float* ab1   = (const float*)d_in[9];
    const float* aw2   = (const float*)d_in[10];
    const float* ab2   = (const float*)d_in[11];
    const int*   eidx  = (const int*)d_in[12];

    const int N = in_sizes[0] / 128;
    const int E = in_sizes[1] / 128;
    const int* srcI = eidx;
    const int* dstI = eidx + E;

    // workspace carve-out (256B aligned slices)
    char*  ws  = (char*)d_ws;
    size_t off = 0;
    auto carve = [&](size_t bytes) -> char* {
        char* p = ws + off;
        off += (bytes + 255) & ~(size_t)255;
        return p;
    };
    float*          msum  = (float*)         carve((size_t)N * 128 * sizeof(float));
    float*          cnt   = (float*)         carve((size_t)N * sizeof(float));
    unsigned short* hbf   = (unsigned short*)carve((size_t)N * 128 * sizeof(unsigned short));
    unsigned short* w1sw  = (unsigned short*)carve((size_t)384 * 128 * sizeof(unsigned short));
    unsigned short* w2sw  = (unsigned short*)carve((size_t)128 * 128 * sizeof(unsigned short));
    unsigned short* aw1sw = (unsigned short*)carve((size_t)256 * 128 * sizeof(unsigned short));
    unsigned short* aw2sw = (unsigned short*)carve((size_t)128 * 128 * sizeof(unsigned short));
    (void)ws_size; (void)n_in; (void)out_size;

    hipMemsetAsync(msum, 0, (size_t)N * 128 * sizeof(float), stream);
    hipMemsetAsync(cnt,  0, (size_t)N * sizeof(float), stream);

    // LayerNorm -> bf16 h
    ln_to_bf16_kernel<<<(N + 7) / 8, 256, 0, stream>>>(node, gamma, beta, hbf, N);

    // Weight swizzles (tiny)
    swizzle_w_kernel<<<(12 * 8 * 32 * 16 + 255) / 256, 256, 0, stream>>>(w1,  w1sw,  12);
    swizzle_w_kernel<<<( 4 * 8 * 32 * 16 + 255) / 256, 256, 0, stream>>>(w2,  w2sw,   4);
    swizzle_w_kernel<<<( 8 * 8 * 32 * 16 + 255) / 256, 256, 0, stream>>>(aw1, aw1sw,  8);
    swizzle_w_kernel<<<( 4 * 8 * 32 * 16 + 255) / 256, 256, 0, stream>>>(aw2, aw2sw,  4);

    // Degree counts
    edge_count_kernel<<<(E + 255) / 256, 256, 0, stream>>>(srcI, cnt, E);

    // Edge MLP + scatter-add
    edge_msg_kernel<<<(E + 15) / 16, 256, 0, stream>>>(
        hbf, ef, srcI, dstI, w1sw, b1, w2sw, b2, msum, E);

    // Node MLP + residual
    node_agg_kernel<<<(N + 15) / 16, 256, 0, stream>>>(
        node, hbf, msum, cnt, aw1sw, ab1, aw2sw, ab2, (float*)d_out, N);
}
